// ChemResBlock_70531952935587
// MI455X (gfx1250) — matrix operand order repaired and checked by reading, not compiled, not run on hardware
//
#include <hip/hip_runtime.h>

// Problem constants (from reference)
#define S_DIM 16
#define A_DIM 512
#define F_DIM 64
#define L_DIM 12
#define KB    792          // L*(F+2)
#define KPAD  800          // padded to multiple of 32

typedef __bf16 bf16;
typedef __attribute__((ext_vector_type(16))) __bf16 v16bf;
typedef __attribute__((ext_vector_type(8)))  float  v8f;
typedef __attribute__((ext_vector_type(4)))  unsigned int v4u;
typedef __attribute__((ext_vector_type(8)))  int v8i;
typedef __attribute__((ext_vector_type(4)))  int v4i;

union Frag16 { v16bf bf; v4u u[2]; };

__device__ __forceinline__ unsigned short f2bf(float f) {
  bf16 h = (bf16)f;
  return __builtin_bit_cast(unsigned short, h);
}

// ---------------------------------------------------------------------------
// TDM: 2D tile load, tile_dim0=64 elems (bf16), tile_dim1=192 rows,
// row stride 512 elems.  lds_off = LDS byte offset, gaddr = global byte addr.
// ---------------------------------------------------------------------------
__device__ __forceinline__ void tdm_load_conn(unsigned lds_off,
                                              unsigned long long gaddr) {
  v4u g0;
  g0.x = 1u;                                             // count=1, user mode
  g0.y = lds_off;                                        // lds_addr [63:32]
  g0.z = (unsigned)(gaddr & 0xFFFFFFFFu);                // global_addr lo
  g0.w = (unsigned)((gaddr >> 32) & 0x01FFFFFFu)         // global_addr [56:32]
       | 0x80000000u;                                    // type=2 ("image")
  v8i g1;
  g1[0] = 0x00010000;            // workgroup_mask=0, data_size=1 (2 bytes)
  g1[1] = (int)(64u << 16);      // tensor_dim0[15:0]=64   (bits 79:48)
  g1[2] = (int)(192u << 16);     // tensor_dim0 hi=0 | tensor_dim1[15:0]=192
  g1[3] = (int)(64u << 16);      // tensor_dim1 hi=0 | tile_dim0=64
  g1[4] = 192;                   // tile_dim1=192, tile_dim2=0
  g1[5] = 512;                   // tensor_dim0_stride lo = 512
  g1[6] = 0;                     // stride hi, tensor_dim1_stride lo
  g1[7] = 0;
  v4i z4 = {0, 0, 0, 0};
#if __clang_major__ >= 23
  v8i z8 = {0, 0, 0, 0, 0, 0, 0, 0};
  __builtin_amdgcn_tensor_load_to_lds(g0, g1, z4, z4, z8, 0);
#else
  __builtin_amdgcn_tensor_load_to_lds(g0, g1, z4, z4, 0);
#endif
}

// ---------------------------------------------------------------------------
// Filters (3,64,12,66) f32 -> wpad (3,64,800) bf16, zero padded 792..799.
// ---------------------------------------------------------------------------
__global__ __launch_bounds__(256)
void prep_filters_k(const float* __restrict__ filt, unsigned short* __restrict__ wpad) {
  int gid = blockIdx.x * 256 + threadIdx.x;        // 3*64*800 total
  int conv = gid / (F_DIM * KPAD);
  int r    = gid - conv * (F_DIM * KPAD);
  int o    = r / KPAD;
  int k    = r - o * KPAD;
  float v = (k < KB) ? filt[((size_t)(conv * F_DIM + o)) * KB + k] : 0.0f;
  wpad[gid] = f2bf(v);
}

// ---------------------------------------------------------------------------
// conn (S,A,A,L) f32 -> connT (S,A,L,A) bf16  (A-matrix rows, K contiguous).
// One block per (s,a): coalesced read, LDS transpose, coalesced write.
// ---------------------------------------------------------------------------
__global__ __launch_bounds__(256)
void prep_connT_k(const float* __restrict__ conn, unsigned short* __restrict__ connT) {
  __shared__ unsigned short t[L_DIM * 514];        // padded stride vs bank conflicts
  const int sa = blockIdx.x;                       // s*512 + a
  const float* src = conn + (size_t)sa * (A_DIM * L_DIM);
  for (int i = threadIdx.x; i < A_DIM * L_DIM; i += 256) {
    int k = i / L_DIM;
    int l = i - k * L_DIM;
    t[l * 514 + k] = f2bf(src[i]);
  }
  __syncthreads();
  unsigned short* dstp = connT + (size_t)sa * (L_DIM * A_DIM);
  for (int i = threadIdx.x; i < A_DIM * L_DIM; i += 256) {
    int l = i >> 9;
    int k = i & 511;
    dstp[i] = t[l * 514 + k];
  }
}

// ---------------------------------------------------------------------------
// Activations x (S,A,F) f32 -> xT (S,F,A) bf16  (B-matrix for phase A).
// ---------------------------------------------------------------------------
__global__ __launch_bounds__(256)
void transpose_x_k(const float* __restrict__ x, unsigned short* __restrict__ xT) {
  int gid  = blockIdx.x * 256 + threadIdx.x;       // S*F*A = 524288
  int k    = gid & (A_DIM - 1);
  int rest = gid >> 9;
  int f    = rest & (F_DIM - 1);
  int s    = rest >> 6;
  float v = x[((size_t)(s * A_DIM + k)) * F_DIM + f];
  xT[gid] = f2bf(v);                                // xT[(s*64+f)*512 + k]
}

// ---------------------------------------------------------------------------
// Fused ChemConv (TDM path):  y = relu( conv(x) [+ node] )
// Phase A: ncv = conn x X (M=192,K=512,N=64), conn tiles DMA'd by the TDM
// into double-buffered LDS.  Phase B: out = comb x W^T (M=16,K=800,N=64),
// residual + relu fused.
// ---------------------------------------------------------------------------
__global__ __launch_bounds__(256)
void chemconv_tdm_k(const unsigned short* __restrict__ connT,
                    const float* __restrict__ bondt,
                    const float* __restrict__ node,
                    const unsigned short* __restrict__ xT,
                    const unsigned short* __restrict__ wpad,
                    float* __restrict__ dst,
                    int mode) {
  __shared__ unsigned short connA[2][192 * 64];  // ping-pong 64-wide K chunks
  __shared__ unsigned short comb[16 * KPAD];     // rows a, K = l*66+c (bf16)

  const int tid  = threadIdx.x;
  const int lane = tid & 31;
  const int wave = tid >> 5;
  const int s    = blockIdx.x >> 5;
  const int a0   = (blockIdx.x & 31) * 16;

  // Prefill bond columns (c=64,65) and zero pad (792..799) of comb.
  for (int i = tid; i < 16 * L_DIM * 2; i += 256) {
    int a = i / (L_DIM * 2);
    int rem = i - a * (L_DIM * 2);
    int l = rem >> 1, b = rem & 1;
    float v = bondt[(((size_t)(s * A_DIM) + a0 + a) * L_DIM + l) * 2 + b];
    comb[a * KPAD + l * 66 + 64 + b] = f2bf(v);
  }
  for (int i = tid; i < 16 * 8; i += 256) {
    int a = i >> 3, p = i & 7;
    comb[a * KPAD + KB + p] = 0;
  }

  const int nt = wave & 3;     // N tile (0..3)
  const int mg = wave >> 2;    // M tile group (0..1), 6 tiles each
  const int m  = lane & 15;
  const int h  = lane >> 4;

  v8f acc[6];
#pragma unroll
  for (int t = 0; t < 6; ++t) acc[t] = (v8f){0.f,0.f,0.f,0.f,0.f,0.f,0.f,0.f};

  const unsigned short* xTs = xT + (size_t)s * F_DIM * A_DIM;

  // Global byte address of this block's 192 A-rows, K=0.
  const unsigned long long gbase =
      (unsigned long long)(size_t)connT +
      (unsigned long long)(s * A_DIM + a0) * (L_DIM * A_DIM) * 2ull;
  const unsigned lds0 = (unsigned)(unsigned long long)(size_t)&connA[0][0];
  const unsigned lds1 = (unsigned)(unsigned long long)(size_t)&connA[1][0];

  if (wave == 0) tdm_load_conn(lds0, gbase);     // prefetch chunk 0

  for (int kc = 0; kc < 8; ++kc) {               // K chunks of 64 over K=512
    if (wave == 0) {
      if (kc < 7) {
        // Prefetch next chunk into the other buffer, then wait so that the
        // current chunk (in-order ahead of it) is complete.
        tdm_load_conn((kc & 1) ? lds0 : lds1,
                      gbase + (unsigned long long)(kc + 1) * 128ull);
        __builtin_amdgcn_s_wait_tensorcnt(1);
      } else {
        __builtin_amdgcn_s_wait_tensorcnt(0);
      }
    }
    __syncthreads();                             // chunk kc visible to all waves
    const unsigned short* Abuf = connA[kc & 1];

#pragma unroll
    for (int ks = 0; ks < 2; ++ks) {             // two K=32 WMMA steps per chunk
      Frag16 bfr;                                // B: x column f, 32 K values
      {
        const unsigned short* p =
            xTs + (size_t)(nt * 16 + m) * A_DIM + kc * 64 + ks * 32 + h * 8;
        bfr.u[0] = *(const v4u*)p;
        bfr.u[1] = *(const v4u*)(p + 16);
      }
#pragma unroll
      for (int t = 0; t < 6; ++t) {
        const int mt = mg * 6 + t;
        Frag16 afr;                              // A: conn row (a,l), 32 K values
        const unsigned short* p = Abuf + (mt * 16 + m) * 64 + ks * 32 + h * 8;
        afr.u[0] = *(const v4u*)p;
        afr.u[1] = *(const v4u*)(p + 16);
        acc[t] = __builtin_amdgcn_wmma_f32_16x16x32_bf16(
            false, afr.bf, false, bfr.bf, (short)0, acc[t], false, false);
      }
    }
    __syncthreads();                             // done reading buf before reuse
  }

  // Scatter phase-A tiles into comb (bf16): row a, col l*66 + f.
  {
    const int hh = h * 8;
#pragma unroll
    for (int t = 0; t < 6; ++t) {
      const int mt = mg * 6 + t;
#pragma unroll
      for (int i = 0; i < 8; ++i) {
        int r = mt * 16 + i + hh;                // global M row = a*12 + l
        int a = r / L_DIM;
        int l = r - a * L_DIM;
        comb[a * KPAD + l * 66 + nt * 16 + m] = f2bf(acc[t][i]);
      }
    }
  }
  __syncthreads();

  // Phase B: out[a,o] = sum_k comb[a,k] * wpad[o,k];  waves 0..3, one N tile each.
  if (wave < 4) {
    v8f acc2 = (v8f){0.f,0.f,0.f,0.f,0.f,0.f,0.f,0.f};
    for (int kk = 0; kk < KPAD / 32; ++kk) {
      Frag16 afr, bfr;
      const unsigned short* pa = &comb[m * KPAD + kk * 32 + h * 8];
      afr.u[0] = *(const v4u*)pa;
      afr.u[1] = *(const v4u*)(pa + 16);
      const unsigned short* pb = wpad + (size_t)(wave * 16 + m) * KPAD + kk * 32 + h * 8;
      bfr.u[0] = *(const v4u*)pb;
      bfr.u[1] = *(const v4u*)(pb + 16);
      acc2 = __builtin_amdgcn_wmma_f32_16x16x32_bf16(
          false, afr.bf, false, bfr.bf, (short)0, acc2, false, false);
    }
    const int hh = h * 8;
#pragma unroll
    for (int i = 0; i < 8; ++i) {
      int a = i + hh;
      int o = wave * 16 + m;
      size_t idx = ((size_t)(s * A_DIM) + a0 + a) * F_DIM + o;
      float y = acc2[i];
      if (mode) y += node[idx];
      dst[idx] = y > 0.f ? y : 0.f;
    }
  }
}

// ---------------------------------------------------------------------------
// Fallback ChemConv (no connT workspace): stages f32 conn via VALU convert.
// ---------------------------------------------------------------------------
__global__ __launch_bounds__(256)
void chemconv_stage_k(const float* __restrict__ conn,
                      const float* __restrict__ bondt,
                      const float* __restrict__ node,
                      const unsigned short* __restrict__ xT,
                      const unsigned short* __restrict__ wpad,
                      float* __restrict__ dst,
                      int mode) {
  __shared__ unsigned short connA[192 * 64];
  __shared__ unsigned short comb[16 * KPAD];

  const int tid  = threadIdx.x;
  const int lane = tid & 31;
  const int wave = tid >> 5;
  const int s    = blockIdx.x >> 5;
  const int a0   = (blockIdx.x & 31) * 16;

  for (int i = tid; i < 16 * L_DIM * 2; i += 256) {
    int a = i / (L_DIM * 2);
    int rem = i - a * (L_DIM * 2);
    int l = rem >> 1, b = rem & 1;
    float v = bondt[(((size_t)(s * A_DIM) + a0 + a) * L_DIM + l) * 2 + b];
    comb[a * KPAD + l * 66 + 64 + b] = f2bf(v);
  }
  for (int i = tid; i < 16 * 8; i += 256) {
    int a = i >> 3, p = i & 7;
    comb[a * KPAD + KB + p] = 0;
  }

  const int nt = wave & 3;
  const int mg = wave >> 2;
  const int m  = lane & 15;
  const int h  = lane >> 4;

  v8f acc[6];
#pragma unroll
  for (int t = 0; t < 6; ++t) acc[t] = (v8f){0.f,0.f,0.f,0.f,0.f,0.f,0.f,0.f};

  const unsigned short* xTs = xT + (size_t)s * F_DIM * A_DIM;

  for (int kc = 0; kc < 8; ++kc) {
    const float* cg = conn + ((size_t)(s * A_DIM + a0) * A_DIM + kc * 64) * L_DIM;
    for (int i = tid; i < 16 * 64 * L_DIM; i += 256) {
      int a   = i / (64 * L_DIM);
      int rem = i - a * (64 * L_DIM);
      float v = cg[(size_t)a * (A_DIM * L_DIM) + rem];
      int kk = rem / L_DIM;
      int l  = rem - kk * L_DIM;
      connA[(a * L_DIM + l) * 64 + kk] = f2bf(v);
    }
    __syncthreads();

#pragma unroll
    for (int ks = 0; ks < 2; ++ks) {
      Frag16 bfr;
      {
        const unsigned short* p =
            xTs + (size_t)(nt * 16 + m) * A_DIM + kc * 64 + ks * 32 + h * 8;
        bfr.u[0] = *(const v4u*)p;
        bfr.u[1] = *(const v4u*)(p + 16);
      }
#pragma unroll
      for (int t = 0; t < 6; ++t) {
        const int mt = mg * 6 + t;
        Frag16 afr;
        const unsigned short* p = &connA[(mt * 16 + m) * 64 + ks * 32 + h * 8];
        afr.u[0] = *(const v4u*)p;
        afr.u[1] = *(const v4u*)(p + 16);
        acc[t] = __builtin_amdgcn_wmma_f32_16x16x32_bf16(
            false, afr.bf, false, bfr.bf, (short)0, acc[t], false, false);
      }
    }
    __syncthreads();
  }

  {
    const int hh = h * 8;
#pragma unroll
    for (int t = 0; t < 6; ++t) {
      const int mt = mg * 6 + t;
#pragma unroll
      for (int i = 0; i < 8; ++i) {
        int r = mt * 16 + i + hh;
        int a = r / L_DIM;
        int l = r - a * L_DIM;
        comb[a * KPAD + l * 66 + nt * 16 + m] = f2bf(acc[t][i]);
      }
    }
  }
  __syncthreads();

  if (wave < 4) {
    v8f acc2 = (v8f){0.f,0.f,0.f,0.f,0.f,0.f,0.f,0.f};
    for (int kk = 0; kk < KPAD / 32; ++kk) {
      Frag16 afr, bfr;
      const unsigned short* pa = &comb[m * KPAD + kk * 32 + h * 8];
      afr.u[0] = *(const v4u*)pa;
      afr.u[1] = *(const v4u*)(pa + 16);
      const unsigned short* pb = wpad + (size_t)(wave * 16 + m) * KPAD + kk * 32 + h * 8;
      bfr.u[0] = *(const v4u*)pb;
      bfr.u[1] = *(const v4u*)(pb + 16);
      acc2 = __builtin_amdgcn_wmma_f32_16x16x32_bf16(
          false, afr.bf, false, bfr.bf, (short)0, acc2, false, false);
    }
    const int hh = h * 8;
#pragma unroll
    for (int i = 0; i < 8; ++i) {
      int a = i + hh;
      int o = wave * 16 + m;
      size_t idx = ((size_t)(s * A_DIM) + a0 + a) * F_DIM + o;
      float y = acc2[i];
      if (mode) y += node[idx];
      dst[idx] = y > 0.f ? y : 0.f;
    }
  }
}

// ---------------------------------------------------------------------------
extern "C" void kernel_launch(void* const* d_in, const int* in_sizes, int n_in,
                              void* d_out, int out_size, void* d_ws, size_t ws_size,
                              hipStream_t stream) {
  const float* node  = (const float*)d_in[0];
  const float* conn  = (const float*)d_in[1];
  const float* bondt = (const float*)d_in[2];
  const float* filt  = (const float*)d_in[3];
  float* out = (float*)d_out;

  // Workspace: wpad | xT | connT (bf16 each)
  const size_t wpad_elems  = (size_t)3 * F_DIM * KPAD;          // 153600
  const size_t xT_elems    = (size_t)S_DIM * F_DIM * A_DIM;     // 524288
  const size_t connT_elems = (size_t)S_DIM * A_DIM * A_DIM * L_DIM;  // 50331648
  unsigned short* wpad  = (unsigned short*)d_ws;
  unsigned short* xT    = wpad + wpad_elems;
  unsigned short* connT = xT + xT_elems;
  const bool use_tdm = ws_size >= (wpad_elems + xT_elems + connT_elems) * 2;

  prep_filters_k<<<(int)(wpad_elems / 256), 256, 0, stream>>>(filt, wpad);
  if (use_tdm)
    prep_connT_k<<<S_DIM * A_DIM, 256, 0, stream>>>(conn, connT);

  const float* xsrc = node;
  for (int j = 0; j < 6; ++j) {
    transpose_x_k<<<(int)(xT_elems / 256), 256, 0, stream>>>(xsrc, xT);
    if (use_tdm)
      chemconv_tdm_k<<<S_DIM * (A_DIM / 16), 256, 0, stream>>>(
          connT, bondt, node, xT, wpad + (size_t)(j >> 1) * F_DIM * KPAD, out, j & 1);
    else
      chemconv_stage_k<<<S_DIM * (A_DIM / 16), 256, 0, stream>>>(
          conn, bondt, node, xT, wpad + (size_t)(j >> 1) * F_DIM * KPAD, out, j & 1);
    xsrc = out;  // conv fully rewrites d_out; next transpose consumes it
  }
}
